// PerformerAttention_38371237823044
// MI455X (gfx1250) — matrix operand order repaired and checked by reading, not compiled, not run on hardware
//
#include <hip/hip_runtime.h>
#include <hip/hip_bf16.h>
#include <hip/hip_fp16.h>

// Problem constants (match reference)
#define BB   4
#define NN   4096
#define DIMD 512
#define HH   8
#define DHH  64
#define FF   256
#define CSZ  128
#define CCH  32                  // N / CS
#define TOK  (BB * NN)           // 16384 tokens
#define QKVN (3 * HH * DHH)      // 1536
#define NCHUNK (BB * HH * CCH)   // 1024 independent chunks

typedef _Float16 h8  __attribute__((ext_vector_type(8)));
typedef _Float16 h16 __attribute__((ext_vector_type(16)));
typedef float    f8  __attribute__((ext_vector_type(8)));

// ---------------------------------------------------------------------------
// WMMA helpers. All GEMMs: C[M][N] = A[M][K] * Bt[N][K]^T  (f16 in, f32 acc).
// A fragment (16x32 f16): lane l -> row m = l&15; kh = (l>>4)*8,
//   v0..3 = K[kh..kh+7], v4..7 = K[16+kh..16+kh+7]  -> two 16B loads.
// B fragment (32x16 f16): lane l -> col n = l&15; kh = (l>>4)*8,
//   v0..7 = K[kh..kh+7], v8..15 = K[16+kh..16+kh+7]  -> two 16B loads.
// C fragment (16x16 f32): lane l -> n = l&15, rows m = 8*(l>>4) + r, r=0..7.
// ---------------------------------------------------------------------------
__device__ __forceinline__ h16 load_a_frag(const _Float16* A, int lda, int kk, int lane) {
    const _Float16* p = A + (size_t)(lane & 15) * lda + kk + ((lane >> 4) << 3);
    h8 lo = *(const h8*)(p);
    h8 hi = *(const h8*)(p + 16);
    h16 r;
#pragma unroll
    for (int i = 0; i < 8; ++i) { r[i] = lo[i]; r[i + 8] = hi[i]; }
    return r;
}

__device__ __forceinline__ h16 load_b_frag(const _Float16* Bt, int ldb, int kk, int lane) {
    const _Float16* p = Bt + (size_t)(lane & 15) * ldb + kk + ((lane >> 4) << 3);
    h8 lo = *(const h8*)(p);
    h8 hi = *(const h8*)(p + 16);
    h16 r;
#pragma unroll
    for (int i = 0; i < 8; ++i) { r[i] = lo[i]; r[i + 8] = hi[i]; }
    return r;
}

__device__ __forceinline__ f8 wmma16(h16 a, h16 b, f8 c) {
    return __builtin_amdgcn_wmma_f32_16x16x32_f16(false, a, false, b, (short)0, c,
                                                  false, false);
}

// ---------------------------------------------------------------------------
// K0: f32 -> f16 conversion
// ---------------------------------------------------------------------------
__global__ void k_cvt(const float* __restrict__ src, _Float16* __restrict__ dst, int n) {
    int i = blockIdx.x * blockDim.x + threadIdx.x;
    if (i < n) dst[i] = (_Float16)src[i];
}

// ---------------------------------------------------------------------------
// K1: qkv = x @ w_qkv^T, scatter into chunk layouts.
//   q -> qh [cid][s][d] (scaled by DH^-0.5),  k -> kh [cid][s][d],
//   v -> vt [cid][d][s] (transposed).
// Register blocking: each wave computes a 32(M) x 64(N) block = 8 C-tiles.
// Per K-step: 2 A-frags + 4 B-frags feed 8 WMMAs.
// Grid: (TOK/32)*(QKVN/64)/8 blocks of 256 threads.
// ---------------------------------------------------------------------------
__global__ void __launch_bounds__(256) k_qkv(const _Float16* __restrict__ xh,
                                             const _Float16* __restrict__ wqh,
                                             _Float16* __restrict__ qh,
                                             _Float16* __restrict__ kh,
                                             _Float16* __restrict__ vt) {
    const int lane = threadIdx.x & 31;
    const int wave = threadIdx.x >> 5;
    const int NB = QKVN / 64;                        // 24 column blocks
    int job = blockIdx.x * 8 + wave;
    int i = job / NB, j = job % NB;
    int row0 = i * 32, col0 = j * 64;

    const _Float16* A  = xh  + (size_t)row0 * DIMD;
    const _Float16* Bt = wqh + (size_t)col0 * DIMD;
    f8 acc[8] = {};                                   // [n4*2 + m]
#pragma unroll 4
    for (int kk = 0; kk < DIMD; kk += 32) {
        h16 a0 = load_a_frag(A, DIMD, kk, lane);
        h16 a1 = load_a_frag(A + (size_t)16 * DIMD, DIMD, kk, lane);
#pragma unroll
        for (int n4 = 0; n4 < 4; ++n4) {
            h16 b = load_b_frag(Bt + (size_t)n4 * 16 * DIMD, DIMD, kk, lane);
            acc[n4 * 2 + 0] = wmma16(a0, b, acc[n4 * 2 + 0]);
            acc[n4 * 2 + 1] = wmma16(a1, b, acc[n4 * 2 + 1]);
        }
    }

    // col0 is 64-aligned -> whole block lies in one of q/k/v and one head.
    int which = col0 >> 9;                // 0=q 1=k 2=v
    int h = (col0 & 511) >> 6;
    int b = row0 >> 12;
    int pos0 = row0 & 4095;
    int c = pos0 >> 7, s0 = pos0 & 127;
    size_t cid = ((size_t)b * HH + h) * CCH + c;

    int nloc = lane & 15, mb = (lane >> 4) << 3;
#pragma unroll
    for (int n4 = 0; n4 < 4; ++n4) {
        int d = n4 * 16 + nloc;
#pragma unroll
        for (int m = 0; m < 2; ++m) {
#pragma unroll
            for (int r = 0; r < 8; ++r) {
                int s = s0 + m * 16 + mb + r;
                float v = acc[n4 * 2 + m][r];
                if (which == 0)      qh[(cid * CSZ + s) * DHH + d] = (_Float16)(v * 0.125f);
                else if (which == 1) kh[(cid * CSZ + s) * DHH + d] = (_Float16)v;
                else                 vt[(cid * DHH + d) * CSZ + s] = (_Float16)v;
            }
        }
    }
}

// ---------------------------------------------------------------------------
// K2: q' / k' = exp(t @ proj^T - rowmax).  One block per (chunk, q|k).
// Each wave owns 16 rows; A-frag reused across 8 N-tiles per pass (2 passes).
// Scores staged in 128KB dynamic LDS for rowmax+exp; k' also produces
// k_sum[f] and is stored transposed [f][s].
// ---------------------------------------------------------------------------
__global__ void __launch_bounds__(256) k_prime(const _Float16* __restrict__ qh,
                                               const _Float16* __restrict__ kh,
                                               const _Float16* __restrict__ pjh,
                                               _Float16* __restrict__ qp,
                                               _Float16* __restrict__ kpt,
                                               float* __restrict__ ksum) {
    extern __shared__ float smem[];
    float* S    = smem;                 // [128][256]
    float* rmax = smem + CSZ * FF;      // [128]

    int bid = blockIdx.x;
    int which = bid & 1;                // 0=q, 1=k
    size_t cid = (size_t)(bid >> 1);
    const _Float16* src = (which ? kh : qh) + cid * CSZ * DHH;

    const int lane = threadIdx.x & 31;
    const int wave = threadIdx.x >> 5;
    const int row0 = wave * 16;
    const int nloc = lane & 15, mb = (lane >> 4) << 3;

#pragma unroll 1
    for (int half = 0; half < 2; ++half) {
        f8 acc[8] = {};
#pragma unroll
        for (int kk = 0; kk < DHH; kk += 32) {
            h16 a = load_a_frag(src + (size_t)row0 * DHH, DHH, kk, lane);
#pragma unroll
            for (int q = 0; q < 8; ++q) {
                h16 b = load_b_frag(pjh + (size_t)(half * 8 + q) * 16 * DHH, DHH, kk, lane);
                acc[q] = wmma16(a, b, acc[q]);
            }
        }
#pragma unroll
        for (int q = 0; q < 8; ++q) {
            int f = (half * 8 + q) * 16 + nloc;
#pragma unroll
            for (int r = 0; r < 8; ++r) S[(row0 + mb + r) * FF + f] = acc[q][r];
        }
    }
    __syncthreads();

    int tid = threadIdx.x;
    if (tid < CSZ) {
        float m = -1e30f;
        for (int f = 0; f < FF; ++f) m = fmaxf(m, S[tid * FF + f]);
        rmax[tid] = m;
    }
    __syncthreads();

    size_t base = cid * CSZ * FF;
    if (which == 0) {
#pragma unroll 4
        for (int e = 0; e < 128; ++e) {
            int idx = tid * 128 + e;
            int s = idx >> 8;
            qp[base + idx] = (_Float16)__expf(S[idx] - rmax[s]);
        }
    } else {
#pragma unroll 4
        for (int e = 0; e < 128; ++e) {
            int idx = tid * 128 + e;
            int s = idx >> 8, f = idx & 255;
            float v = __expf(S[idx] - rmax[s]);
            kpt[base + (size_t)f * CSZ + s] = (_Float16)v;
            S[idx] = v;
        }
        __syncthreads();
        float acc = 0.f;
        for (int s = 0; s < CSZ; ++s) acc += S[s * FF + tid];
        ksum[cid * FF + tid] = acc;
    }
}

// ---------------------------------------------------------------------------
// K3: ctxT[d][f] = sum_s v[s][d] * k'[s][f].  A = vt [d][s], Bt = kpt [f][s].
// One block per chunk. Wave w: M-tile mi = w>>1 fixed, 8 N-tiles; A-frag
// loaded once per K-step and shared by 8 WMMAs.
// ---------------------------------------------------------------------------
__global__ void __launch_bounds__(256) k_ctx(const _Float16* __restrict__ vt,
                                             const _Float16* __restrict__ kpt,
                                             _Float16* __restrict__ ctxt) {
    size_t cid = blockIdx.x;
    const int lane = threadIdx.x & 31;
    const int wave = threadIdx.x >> 5;
    const _Float16* A  = vt   + cid * DHH * CSZ;
    const _Float16* Bt = kpt  + cid * FF  * CSZ;
    _Float16*       Ct = ctxt + cid * DHH * FF;
    const int nloc = lane & 15, mb = (lane >> 4) << 3;

    int mi = wave >> 1;                   // 0..3
    int nbase = (wave & 1) * 8;           // 0 or 8

    f8 acc[8] = {};
#pragma unroll
    for (int kk = 0; kk < CSZ; kk += 32) {
        h16 a = load_a_frag(A + (size_t)mi * 16 * CSZ, CSZ, kk, lane);
#pragma unroll
        for (int q = 0; q < 8; ++q) {
            h16 b = load_b_frag(Bt + (size_t)(nbase + q) * 16 * CSZ, CSZ, kk, lane);
            acc[q] = wmma16(a, b, acc[q]);
        }
    }
#pragma unroll
    for (int q = 0; q < 8; ++q) {
        int f = (nbase + q) * 16 + nloc;
#pragma unroll
        for (int r = 0; r < 8; ++r)
            Ct[(size_t)(mi * 16 + mb + r) * FF + f] = (_Float16)acc[q][r];
    }
}

// ---------------------------------------------------------------------------
// K4: out[s][d] = (q' @ ctx)[s][d] / (q'[s]·k_sum + 1e-4); store to attn-out
// [token][h*64+d] (f16, A-operand of the final GEMM). A-frag shared by the
// 4 N-tiles per K-step.
// ---------------------------------------------------------------------------
__global__ void __launch_bounds__(256) k_out(const _Float16* __restrict__ qp,
                                             const _Float16* __restrict__ ctxt,
                                             const float* __restrict__ ksum,
                                             _Float16* __restrict__ ao) {
    __shared__ float ks[FF];
    __shared__ float dinv[CSZ];
    size_t cid = blockIdx.x;
    int tid = threadIdx.x;

    ks[tid & (FF - 1)] = ksum[cid * FF + (tid & (FF - 1))];
    __syncthreads();

    const _Float16* Q = qp + cid * CSZ * FF;
    if (tid < CSZ) {
        float acc = 0.f;
        for (int f = 0; f < FF; ++f) acc += (float)Q[(size_t)tid * FF + f] * ks[f];
        dinv[tid] = 1.0f / (acc + 1e-4f);
    }
    __syncthreads();

    int c = (int)(cid & 31);
    int bh = (int)(cid >> 5);
    int h = bh & 7, b = bh >> 3;

    const _Float16* Bt = ctxt + cid * DHH * FF;
    const int lane = threadIdx.x & 31;
    const int wave = threadIdx.x >> 5;
    const int row0 = wave * 16;
    const int nloc = lane & 15, mb = (lane >> 4) << 3;

    f8 acc[4] = {};
#pragma unroll 4
    for (int kk = 0; kk < FF; kk += 32) {
        h16 a = load_a_frag(Q + (size_t)row0 * FF, FF, kk, lane);
#pragma unroll
        for (int nt = 0; nt < 4; ++nt) {
            h16 bf = load_b_frag(Bt + (size_t)nt * 16 * FF, FF, kk, lane);
            acc[nt] = wmma16(a, bf, acc[nt]);
        }
    }
#pragma unroll
    for (int nt = 0; nt < 4; ++nt) {
        int d = nt * 16 + nloc;
#pragma unroll
        for (int r = 0; r < 8; ++r) {
            int s = row0 + mb + r;
            size_t tok = (size_t)b * NN + (size_t)c * CSZ + s;
            ao[tok * (HH * DHH) + h * DHH + d] = (_Float16)(acc[nt][r] * dinv[s]);
        }
    }
}

// ---------------------------------------------------------------------------
// K5: y = ao @ w_out^T + b_out  (f32 output). Same 32x64 register blocking
// as K1.
// ---------------------------------------------------------------------------
__global__ void __launch_bounds__(256) k_final(const _Float16* __restrict__ ao,
                                               const _Float16* __restrict__ woh,
                                               const float* __restrict__ bout,
                                               float* __restrict__ y) {
    const int lane = threadIdx.x & 31;
    const int wave = threadIdx.x >> 5;
    const int NB = DIMD / 64;                         // 8 column blocks
    int job = blockIdx.x * 8 + wave;
    int i = job / NB, j = job % NB;
    int row0 = i * 32, col0 = j * 64;

    const _Float16* A  = ao  + (size_t)row0 * DIMD;
    const _Float16* Bt = woh + (size_t)col0 * DIMD;
    f8 acc[8] = {};
#pragma unroll 4
    for (int kk = 0; kk < DIMD; kk += 32) {
        h16 a0 = load_a_frag(A, DIMD, kk, lane);
        h16 a1 = load_a_frag(A + (size_t)16 * DIMD, DIMD, kk, lane);
#pragma unroll
        for (int n4 = 0; n4 < 4; ++n4) {
            h16 b = load_b_frag(Bt + (size_t)n4 * 16 * DIMD, DIMD, kk, lane);
            acc[n4 * 2 + 0] = wmma16(a0, b, acc[n4 * 2 + 0]);
            acc[n4 * 2 + 1] = wmma16(a1, b, acc[n4 * 2 + 1]);
        }
    }

    int nloc = lane & 15, mb = (lane >> 4) << 3;
#pragma unroll
    for (int n4 = 0; n4 < 4; ++n4) {
        int col = col0 + n4 * 16 + nloc;
        float bias = bout[col];
#pragma unroll
        for (int m = 0; m < 2; ++m) {
#pragma unroll
            for (int r = 0; r < 8; ++r)
                y[(size_t)(row0 + m * 16 + mb + r) * DIMD + col] = acc[n4 * 2 + m][r] + bias;
        }
    }
}

// ---------------------------------------------------------------------------
extern "C" void kernel_launch(void* const* d_in, const int* in_sizes, int n_in,
                              void* d_out, int out_size, void* d_ws, size_t ws_size,
                              hipStream_t stream) {
    const float* x     = (const float*)d_in[0];
    const float* w_qkv = (const float*)d_in[1];
    const float* w_out = (const float*)d_in[2];
    const float* b_out = (const float*)d_in[3];
    const float* proj  = (const float*)d_in[4];
    float* y = (float*)d_out;

    char* ws = (char*)d_ws;
    size_t off = 0;
    auto carve = [&](size_t bytes) {
        void* p = ws + off;
        off = (off + bytes + 255) & ~(size_t)255;
        return p;
    };

    _Float16* xh  = (_Float16*)carve((size_t)TOK * DIMD * 2);
    _Float16* wqh = (_Float16*)carve((size_t)QKVN * DIMD * 2);
    _Float16* woh = (_Float16*)carve((size_t)DIMD * DIMD * 2);
    _Float16* pjh = (_Float16*)carve((size_t)FF * DHH * 2);
    _Float16* qh  = (_Float16*)carve((size_t)NCHUNK * CSZ * DHH * 2);
    _Float16* kh  = (_Float16*)carve((size_t)NCHUNK * CSZ * DHH * 2);
    _Float16* vt  = (_Float16*)carve((size_t)NCHUNK * DHH * CSZ * 2);
    _Float16* qp  = (_Float16*)carve((size_t)NCHUNK * CSZ * FF * 2);
    _Float16* kpt = (_Float16*)carve((size_t)NCHUNK * FF * CSZ * 2);
    float*    ksm = (float*)   carve((size_t)NCHUNK * FF * 4);
    _Float16* ctx = (_Float16*)carve((size_t)NCHUNK * DHH * FF * 2);
    _Float16* ao  = (_Float16*)carve((size_t)TOK * (HH * DHH) * 2);

    // K0: convert inputs to f16
    k_cvt<<<(TOK * DIMD + 255) / 256, 256, 0, stream>>>(x, xh, TOK * DIMD);
    k_cvt<<<(QKVN * DIMD + 255) / 256, 256, 0, stream>>>(w_qkv, wqh, QKVN * DIMD);
    k_cvt<<<(DIMD * DIMD + 255) / 256, 256, 0, stream>>>(w_out, woh, DIMD * DIMD);
    k_cvt<<<(FF * DHH + 255) / 256, 256, 0, stream>>>(proj, pjh, FF * DHH);

    // K1: qkv projection GEMM (32x64 blocks/wave) + scatter to chunk layouts
    k_qkv<<<(TOK / 32) * (QKVN / 64) / 8, 256, 0, stream>>>(xh, wqh, qh, kh, vt);

    // K2: feature maps q', k' (+ k_sum); 128KB + 512B dynamic LDS
    k_prime<<<NCHUNK * 2, 256, (CSZ * FF + CSZ) * sizeof(float), stream>>>(
        qh, kh, pjh, qp, kpt, ksm);

    // K3: per-chunk context GEMM (A-frag reuse x8)
    k_ctx<<<NCHUNK, 256, 0, stream>>>(vt, kpt, ctx);

    // K4: per-chunk output GEMM with D^-1 row rescale (A-frag reuse x4)
    k_out<<<NCHUNK, 256, 0, stream>>>(qp, ctx, ksm, ao);

    // K5: output projection GEMM + bias -> f32 result (32x64 blocks/wave)
    k_final<<<(TOK / 32) * (DIMD / 64) / 8, 256, 0, stream>>>(ao, woh, b_out, y);
}
